// LlamaAttention_68530498175233
// MI455X (gfx1250) — matrix operand order repaired and checked by reading, not compiled, not run on hardware
//
#include <hip/hip_runtime.h>
#include <hip/hip_bf16.h>

typedef _Float16 v16h __attribute__((ext_vector_type(16)));
typedef _Float16 v8h  __attribute__((ext_vector_type(8)));
typedef float    v8f  __attribute__((ext_vector_type(8)));

#define B_  2
#define S_  1024
#define H_  2048
#define NH_ 32
#define NKV_ 8
#define HD_ 64
#define NREP_ (NH_/NKV_)

__device__ __forceinline__ v8f wmma16(v16h a, v16h b, v8f c) {
    return __builtin_amdgcn_wmma_f32_16x16x32_f16(false, a, false, b, (short)0, c, false, false);
}

__device__ __forceinline__ v16h pack16(const _Float16* __restrict__ p) {
    v8h lo = *(const v8h*)p;
    v8h hi = *(const v8h*)(p + 8);
    v16h a;
    #pragma unroll
    for (int i = 0; i < 8; i++) { a[i] = lo[i]; a[i + 8] = hi[i]; }
    return a;
}

// ---------------- reductions ----------------
__global__ void minmax_partial(const float* __restrict__ x, size_t n, float* __restrict__ part) {
    __shared__ float smn[256], smx[256];
    float mn = 0.f, mx = 0.f; // reference does min(.,0)/max(.,0)
    for (size_t i = (size_t)blockIdx.x * blockDim.x + threadIdx.x; i < n;
         i += (size_t)gridDim.x * blockDim.x) {
        float v = x[i];
        mn = fminf(mn, v); mx = fmaxf(mx, v);
    }
    smn[threadIdx.x] = mn; smx[threadIdx.x] = mx;
    __syncthreads();
    for (int s = 128; s > 0; s >>= 1) {
        if ((int)threadIdx.x < s) {
            smn[threadIdx.x] = fminf(smn[threadIdx.x], smn[threadIdx.x + s]);
            smx[threadIdx.x] = fmaxf(smx[threadIdx.x], smx[threadIdx.x + s]);
        }
        __syncthreads();
    }
    if (threadIdx.x == 0) { part[2*blockIdx.x] = smn[0]; part[2*blockIdx.x+1] = smx[0]; }
}

__global__ void minmax_final(const float* __restrict__ part, int nb, float* __restrict__ sc) {
    __shared__ float smn[256], smx[256];
    float mn = 0.f, mx = 0.f;
    for (int i = threadIdx.x; i < nb; i += 256) {
        mn = fminf(mn, part[2*i]); mx = fmaxf(mx, part[2*i+1]);
    }
    smn[threadIdx.x] = mn; smx[threadIdx.x] = mx;
    __syncthreads();
    for (int s = 128; s > 0; s >>= 1) {
        if ((int)threadIdx.x < s) {
            smn[threadIdx.x] = fminf(smn[threadIdx.x], smn[threadIdx.x + s]);
            smx[threadIdx.x] = fmaxf(smx[threadIdx.x], smx[threadIdx.x + s]);
        }
        __syncthreads();
    }
    if (threadIdx.x == 0) {
        float scale = fmaxf((smx[0] - smn[0]) / 65535.f, 1e-12f);
        sc[0] = scale;
        sc[1] = rintf(-smn[0] / scale);
    }
}

__global__ void amax_partial(const float* __restrict__ x, size_t n, float* __restrict__ part) {
    __shared__ float sa[256];
    float a = 0.f;
    for (size_t i = (size_t)blockIdx.x * blockDim.x + threadIdx.x; i < n;
         i += (size_t)gridDim.x * blockDim.x)
        a = fmaxf(a, fabsf(x[i]));
    sa[threadIdx.x] = a;
    __syncthreads();
    for (int s = 128; s > 0; s >>= 1) {
        if ((int)threadIdx.x < s) sa[threadIdx.x] = fmaxf(sa[threadIdx.x], sa[threadIdx.x + s]);
        __syncthreads();
    }
    if (threadIdx.x == 0) part[blockIdx.x] = sa[0];
}

__global__ void amax_final(const float* __restrict__ part, int nb, float* __restrict__ sc, int slot) {
    __shared__ float sa[256];
    float a = 0.f;
    for (int i = threadIdx.x; i < nb; i += 256) a = fmaxf(a, part[i]);
    sa[threadIdx.x] = a;
    __syncthreads();
    for (int s = 128; s > 0; s >>= 1) {
        if ((int)threadIdx.x < s) sa[threadIdx.x] = fmaxf(sa[threadIdx.x], sa[threadIdx.x + s]);
        __syncthreads();
    }
    if (threadIdx.x == 0) sc[slot] = fmaxf(sa[0] / 127.f, 1e-12f);
}

// ---------------- quantize / dequantize ----------------
__global__ void quant16_to_f16(const float* __restrict__ x, _Float16* __restrict__ y,
                               const float* __restrict__ sc, size_t n) {
    size_t i = (size_t)blockIdx.x * blockDim.x + threadIdx.x;
    if (i >= n) return;
    float scale = sc[0], zp = sc[1];
    float q = rintf(x[i] / scale) + zp;
    q = fminf(fmaxf(q, 0.f), 65535.f);
    y[i] = (_Float16)((q - zp) * scale);
}

// LPBQ: per (row, 32-wide block) symmetric 4-bit, one wave per block
__global__ void dequant_w4(const float* __restrict__ w, _Float16* __restrict__ wq, int O, int I) {
    int group = blockIdx.x * (blockDim.x >> 5) + (threadIdx.x >> 5);
    int lane = threadIdx.x & 31;
    int nblk = I / 32;
    if (group >= O * nblk) return;
    int o = group / nblk, blk = group % nblk;
    size_t idx = (size_t)o * I + blk * 32 + lane;
    float x = w[idx];
    float a = fabsf(x);
    #pragma unroll
    for (int m = 16; m >= 1; m >>= 1) a = fmaxf(a, __shfl_xor(a, m, 32));
    float scale = fmaxf(a / 7.f, 1e-12f);
    float q = fminf(fmaxf(rintf(x / scale), -8.f), 7.f);
    wq[idx] = (_Float16)(q * scale);
}

// ---------------- WMMA GEMM: C[M,N] = A[M,K] * W[N,K]^T ----------------
// 32x64 tile per wave: 2 A-frags x 4 B-frags -> 8 WMMAs per 32-wide k step.
__global__ void __launch_bounds__(256)
gemm_wmma(const _Float16* __restrict__ A, const _Float16* __restrict__ W,
          float* __restrict__ C, int M, int N, int K) {
    int wave = blockIdx.x * (blockDim.x >> 5) + (threadIdx.x >> 5);
    int lane = threadIdx.x & 31;
    int ntn = N >> 6;                     // 64-wide N tiles
    if (wave >= (M >> 5) * ntn) return;   // 32-tall M tiles
    int tm = wave / ntn, tn = wave % ntn;
    int m = lane & 15, hlf = lane >> 4;
    const _Float16* arow0 = A + (size_t)(tm * 32 + m) * K + hlf * 16;
    const _Float16* arow1 = arow0 + (size_t)16 * K;
    const _Float16* brow  = W + (size_t)(tn * 64 + m) * K + hlf * 16;
    v8f acc[2][4] = {};
    for (int kk = 0; kk < K; kk += 32) {
        __builtin_prefetch(arow0 + kk + 512, 0, 3);
        __builtin_prefetch(arow1 + kk + 512, 0, 3);
        __builtin_prefetch(brow  + kk + 512, 0, 3);
        v16h a0 = pack16(arow0 + kk);
        v16h a1 = pack16(arow1 + kk);
        #pragma unroll
        for (int j = 0; j < 4; j++) {
            v16h b = *(const v16h*)(brow + (size_t)j * 16 * K + kk);
            acc[0][j] = wmma16(a0, b, acc[0][j]);
            acc[1][j] = wmma16(a1, b, acc[1][j]);
        }
    }
    #pragma unroll
    for (int i = 0; i < 2; i++)
        #pragma unroll
        for (int j = 0; j < 4; j++)
            #pragma unroll
            for (int r = 0; r < 8; r++) {
                int row = tm * 32 + i * 16 + r + hlf * 8;
                int col = tn * 64 + j * 16 + m;
                C[(size_t)row * N + col] = acc[i][j][r];
            }
}

// ---------------- RoPE (+ layout transform to (B,H,S,HD)) ----------------
__global__ void rope_q_f16(const float* __restrict__ qb, const float* __restrict__ cb,
                           const float* __restrict__ sb, _Float16* __restrict__ qr) {
    size_t idx = (size_t)blockIdx.x * blockDim.x + threadIdx.x;
    size_t total = (size_t)B_ * S_ * NH_ * HD_;
    if (idx >= total) return;
    int d = idx % HD_;
    int h = (idx / HD_) % NH_;
    int s = (idx / ((size_t)HD_ * NH_)) % S_;
    int b = idx / ((size_t)HD_ * NH_ * S_);
    size_t base = ((size_t)(b * S_ + s) * NH_ + h) * HD_;
    float x = qb[base + d];
    float rot = (d < HD_/2) ? -qb[base + d + HD_/2] : qb[base + d - HD_/2];
    size_t cbase = ((size_t)b * S_ + s) * HD_ + d;
    float y = x * cb[cbase] + rot * sb[cbase];
    qr[(((size_t)b * NH_ + h) * S_ + s) * HD_ + d] = (_Float16)y;
}

__global__ void rope_k_f32(const float* __restrict__ kb, const float* __restrict__ cb,
                           const float* __restrict__ sb, float* __restrict__ kr) {
    size_t idx = (size_t)blockIdx.x * blockDim.x + threadIdx.x;
    size_t total = (size_t)B_ * S_ * NKV_ * HD_;
    if (idx >= total) return;
    int d = idx % HD_;
    int h = (idx / HD_) % NKV_;
    int s = (idx / ((size_t)HD_ * NKV_)) % S_;
    int b = idx / ((size_t)HD_ * NKV_ * S_);
    size_t base = ((size_t)(b * S_ + s) * NKV_ + h) * HD_;
    float x = kb[base + d];
    float rot = (d < HD_/2) ? -kb[base + d + HD_/2] : kb[base + d - HD_/2];
    size_t cbase = ((size_t)b * S_ + s) * HD_ + d;
    float y = x * cb[cbase] + rot * sb[cbase];
    kr[(((size_t)b * NKV_ + h) * S_ + s) * HD_ + d] = y;
}

// fq8_sym cast, same layout (K path, already (B,NKV,S,HD))
__global__ void quant8_copy(const float* __restrict__ x, _Float16* __restrict__ y,
                            const float* __restrict__ sc, int slot, size_t n) {
    size_t i = (size_t)blockIdx.x * blockDim.x + threadIdx.x;
    if (i >= n) return;
    float s = sc[slot];
    float q = fminf(fmaxf(rintf(x[i] / s), -128.f), 127.f);
    y[i] = (_Float16)(q * s);
}

// fq8_sym cast for V, transposed to (B,NKV,HD,S) so PV B-frags load contiguously
__global__ void quant8_v_tr(const float* __restrict__ vb, _Float16* __restrict__ vt,
                            const float* __restrict__ sc, int slot) {
    size_t idx = (size_t)blockIdx.x * blockDim.x + threadIdx.x;
    size_t total = (size_t)B_ * S_ * NKV_ * HD_;
    if (idx >= total) return;
    int d = idx % HD_;
    int h = (idx / HD_) % NKV_;
    int s = (idx / ((size_t)HD_ * NKV_)) % S_;
    int b = idx / ((size_t)HD_ * NKV_ * S_);
    float s8 = sc[slot];
    float x = vb[idx];  // (B,S,NKV,HD) contiguous read
    float q = fminf(fmaxf(rintf(x / s8), -128.f), 127.f);
    vt[(((size_t)b * NKV_ + h) * HD_ + d) * S_ + s] = (_Float16)(q * s8);
}

// ---------------- fused attention: 16 query rows per block, 4 waves ----------------
__global__ void __launch_bounds__(128)
attn_kernel(const _Float16* __restrict__ qr, const _Float16* __restrict__ kr,
            const _Float16* __restrict__ vt, float* __restrict__ attn_out,
            float* __restrict__ ctx, float scaling) {
    extern __shared__ char smem[];
    float*    scores = (float*)smem;                         // 16*S f32
    _Float16* probs  = (_Float16*)(smem + 16 * S_ * 4);      // 16*S f16
    float*    red    = (float*)(smem + 16 * S_ * 4 + 16 * S_ * 2); // 256 + 32 f32

    int blk = blockIdx.x;
    int qt = blk % (S_ / 16);
    int h  = (blk / (S_ / 16)) % NH_;
    int b  = blk / ((S_ / 16) * NH_);
    int hkv = h / NREP_;
    int tid = threadIdx.x, lane = tid & 31, wave = tid >> 5;
    int m = lane & 15, hlf = lane >> 4;

    const _Float16* qbase  = qr + (size_t)(b * NH_  + h)   * S_ * HD_;
    const _Float16* kbase  = kr + (size_t)(b * NKV_ + hkv) * S_ * HD_;
    const _Float16* vtbase = vt + (size_t)(b * NKV_ + hkv) * HD_ * S_;

    // Q A-frags are loop-invariant across key tiles: hoist.
    v16h aq[2];
    #pragma unroll
    for (int step = 0; step < 2; step++)
        aq[step] = pack16(qbase + (size_t)(qt * 16 + m) * HD_ + step * 32 + hlf * 16);

    // --- scores = scaling * Q K^T, 16x16 tiles across S, 2 WMMA per tile (HD=64) ---
    for (int kt = wave; kt < S_ / 16; kt += 4) {
        v8f acc = {};
        #pragma unroll
        for (int step = 0; step < 2; step++) {
            const _Float16* bp = kbase + (size_t)(kt * 16 + m) * HD_ + step * 32 + hlf * 16;
            v16h bf = *(const v16h*)bp;
            acc = wmma16(aq[step], bf, acc);
        }
        #pragma unroll
        for (int r = 0; r < 8; r++)
            scores[(r + hlf * 8) * S_ + kt * 16 + m] = acc[r] * scaling;
    }
    __syncthreads();

    // --- softmax with (rowmin - 20) masking; 8 threads per row for stats ---
    int row = tid >> 3, sub = tid & 7;
    int qglob = qt * 16 + row;
    float mn = 3.4e38f, mxa = -3.4e38f;
    for (int k0 = sub; k0 < S_; k0 += 8) {
        float v = scores[row * S_ + k0];
        mn = fminf(mn, v);
        if (k0 <= qglob) mxa = fmaxf(mxa, v);
    }
    red[row * 8 + sub] = mn;
    red[128 + row * 8 + sub] = mxa;
    __syncthreads();
    float* rowstat = red + 256;
    if (tid < 16) {
        float a = 3.4e38f, c = -3.4e38f;
        for (int i = 0; i < 8; i++) {
            a = fminf(a, red[tid * 8 + i]);
            c = fmaxf(c, red[128 + tid * 8 + i]);
        }
        rowstat[tid * 2]     = a;                       // rowmin
        rowstat[tid * 2 + 1] = fmaxf(c, a - 20.f);      // rowmax after masking
    }
    __syncthreads();
    float rmn = rowstat[row * 2], rmx = rowstat[row * 2 + 1];
    float psum = 0.f;
    for (int k0 = sub; k0 < S_; k0 += 8) {
        float v = scores[row * S_ + k0];
        v = (k0 <= qglob) ? v : (rmn - 20.f);
        float e = __expf(v - rmx);
        scores[row * S_ + k0] = e;
        psum += e;
    }
    __syncthreads();
    red[row * 8 + sub] = psum;
    __syncthreads();
    if (tid < 16) {
        float ssum = 0.f;
        for (int i = 0; i < 8; i++) ssum += red[tid * 8 + i];
        rowstat[tid * 2] = 1.f / ssum;
    }
    __syncthreads();
    // --- normalize + emit probs: linear sweep so global stores coalesce (128B/wave) ---
    float* aoutb = attn_out + ((size_t)(b * NH_ + h) * S_ + (size_t)qt * 16) * S_;
    for (int idx = tid; idx < 16 * S_; idx += 128) {
        int r2 = idx >> 10;            // S_ == 1024
        float p = scores[idx] * rowstat[r2 * 2];
        probs[idx] = (_Float16)p;
        aoutb[(size_t)r2 * S_ + (idx & (S_ - 1))] = p;   // (B,NH,S,S) prob output
    }
    __syncthreads();

    // --- PV: each wave one 16x16 output tile (wave -> hd columns), V^T contiguous ---
    v8f acc = {};
    const _Float16* vrow = vtbase + (size_t)(wave * 16 + m) * S_ + hlf * 16;
    for (int step = 0; step < S_ / 32; step++) {
        v16h a = pack16(probs + (size_t)m * S_ + step * 32 + hlf * 16);
        v16h bf = *(const v16h*)(vrow + step * 32);
        acc = wmma16(a, bf, acc);
    }
    #pragma unroll
    for (int r = 0; r < 8; r++) {
        int rrow = qt * 16 + r + hlf * 8;
        ctx[((size_t)(b * NH_ + h) * S_ + rrow) * HD_ + wave * 16 + m] = acc[r];
    }
}

// (B,NH,S,HD) f32 -> (B,S,NH*HD) f16 for the output projection
__global__ void ctx_to_f16(const float* __restrict__ ctx, _Float16* __restrict__ out) {
    size_t idx = (size_t)blockIdx.x * blockDim.x + threadIdx.x;
    size_t total = (size_t)B_ * NH_ * S_ * HD_;
    if (idx >= total) return;
    int d = idx % HD_;
    int s = (idx / HD_) % S_;
    int h = (idx / ((size_t)HD_ * S_)) % NH_;
    int b = idx / ((size_t)HD_ * S_ * NH_);
    out[((size_t)(b * S_ + s) * NH_ + h) * HD_ + d] = (_Float16)ctx[idx];
}

extern "C" void kernel_launch(void* const* d_in, const int* in_sizes, int n_in,
                              void* d_out, int out_size, void* d_ws, size_t ws_size,
                              hipStream_t stream) {
    const float* hidden = (const float*)d_in[0];
    const float* cosb   = (const float*)d_in[1];
    const float* sinb   = (const float*)d_in[2];
    // d_in[3] = attention_mask (int32 causal) -- equivalent to key<=query, applied in-kernel
    const float* q_w    = (const float*)d_in[4];
    const float* k_w    = (const float*)d_in[5];
    const float* v_w    = (const float*)d_in[6];
    const float* o_w    = (const float*)d_in[7];

    char* ws = (char*)d_ws;
    size_t off = 0;
    auto alloc = [&](size_t bytes) { char* p = ws + off; off += (bytes + 255) & ~(size_t)255; return p; };

    float*    part = (float*)alloc(8192);
    float*    sc   = (float*)alloc(256);   // [0]=scale16 [1]=zp16 [2]=kscale [3]=vscale
    _Float16* hq   = (_Float16*)alloc((size_t)B_*S_*H_*2);
    _Float16* wqq  = (_Float16*)alloc((size_t)NH_*HD_*H_*2);
    _Float16* wqk  = (_Float16*)alloc((size_t)NKV_*HD_*H_*2);
    _Float16* wqv  = (_Float16*)alloc((size_t)NKV_*HD_*H_*2);
    _Float16* wqo  = (_Float16*)alloc((size_t)H_*NH_*HD_*2);
    float*    qbuf = (float*)alloc((size_t)B_*S_*NH_*HD_*4);
    float*    kbuf = (float*)alloc((size_t)B_*S_*NKV_*HD_*4);
    float*    vbuf = (float*)alloc((size_t)B_*S_*NKV_*HD_*4);
    _Float16* qr   = (_Float16*)alloc((size_t)B_*NH_*S_*HD_*2);
    float*    krf  = (float*)alloc((size_t)B_*NKV_*S_*HD_*4);
    _Float16* kr   = (_Float16*)alloc((size_t)B_*NKV_*S_*HD_*2);
    _Float16* vt   = (_Float16*)alloc((size_t)B_*NKV_*HD_*S_*2);
    float*    ctxf = (float*)alloc((size_t)B_*NH_*S_*HD_*4);
    _Float16* ctxh = (_Float16*)alloc((size_t)B_*S_*NH_*HD_*2);

    float* out_proj = (float*)d_out;                        // (B,S,H)
    float* out_attn = (float*)d_out + (size_t)B_*S_*H_;     // (B,NH,S,S)

    const size_t nHid = (size_t)B_ * S_ * H_;

    // 1) fq16 params of hidden_states
    minmax_partial<<<256, 256, 0, stream>>>(hidden, nHid, part);
    minmax_final<<<1, 256, 0, stream>>>(part, 256, sc);
    // 2) fq16 -> f16 activations
    quant16_to_f16<<<(int)((nHid + 255) / 256), 256, 0, stream>>>(hidden, hq, sc, nHid);
    // 3) LPBQ dequant weights -> f16
    dequant_w4<<<(NH_*HD_ * (H_/32)) / 8, 256, 0, stream>>>(q_w, wqq, NH_*HD_, H_);
    dequant_w4<<<(NKV_*HD_ * (H_/32)) / 8, 256, 0, stream>>>(k_w, wqk, NKV_*HD_, H_);
    dequant_w4<<<(NKV_*HD_ * (H_/32)) / 8, 256, 0, stream>>>(v_w, wqv, NKV_*HD_, H_);
    dequant_w4<<<(H_ * (NH_*HD_/32)) / 8, 256, 0, stream>>>(o_w, wqo, H_, NH_*HD_);
    // 4) QKV projections (WMMA, 32x64 tiles)
    {
        int M = B_ * S_;
        int tiles_q  = (M/32) * ((NH_*HD_)/64);
        int tiles_kv = (M/32) * ((NKV_*HD_)/64);
        gemm_wmma<<<(tiles_q  + 7) / 8, 256, 0, stream>>>(hq, wqq, qbuf, M, NH_*HD_, H_);
        gemm_wmma<<<(tiles_kv + 7) / 8, 256, 0, stream>>>(hq, wqk, kbuf, M, NKV_*HD_, H_);
        gemm_wmma<<<(tiles_kv + 7) / 8, 256, 0, stream>>>(hq, wqv, vbuf, M, NKV_*HD_, H_);
    }
    // 5) RoPE + quantize K/V
    {
        size_t nq = (size_t)B_*S_*NH_*HD_, nk = (size_t)B_*S_*NKV_*HD_;
        rope_q_f16<<<(int)((nq + 255) / 256), 256, 0, stream>>>(qbuf, cosb, sinb, qr);
        rope_k_f32<<<(int)((nk + 255) / 256), 256, 0, stream>>>(kbuf, cosb, sinb, krf);
        amax_partial<<<256, 256, 0, stream>>>(krf, nk, part);
        amax_final<<<1, 256, 0, stream>>>(part, 256, sc, 2);
        quant8_copy<<<(int)((nk + 255) / 256), 256, 0, stream>>>(krf, kr, sc, 2, nk);
        amax_partial<<<256, 256, 0, stream>>>(vbuf, nk, part);
        amax_final<<<1, 256, 0, stream>>>(part, 256, sc, 3);
        quant8_v_tr<<<(int)((nk + 255) / 256), 256, 0, stream>>>(vbuf, vt, sc, 3);
    }
    // 6) fused attention (scores LDS 64KB + probs 32KB + reductions)
    {
        int blocks = B_ * NH_ * (S_ / 16);
        size_t shmem = (size_t)16 * S_ * 4 + (size_t)16 * S_ * 2 + (256 + 32) * 4;
        attn_kernel<<<blocks, 128, shmem, stream>>>(qr, kr, vt, out_attn, ctxf, 0.125f);
    }
    // 7) transpose context + output projection straight into d_out
    {
        size_t nc = (size_t)B_ * NH_ * S_ * HD_;
        ctx_to_f16<<<(int)((nc + 255) / 256), 256, 0, stream>>>(ctxf, ctxh);
        int M = B_ * S_;
        int tiles = (M/32) * (H_/64);
        gemm_wmma<<<(tiles + 7) / 8, 256, 0, stream>>>(ctxh, wqo, out_proj, M, H_, H_);
    }
    (void)in_sizes; (void)n_in; (void)out_size; (void)ws_size;
}
